// MangaInpaintGenerator_66005057405595
// MI455X (gfx1250) — matrix operand order repaired and checked by reading, not compiled, not run on hardware
//
#include <hip/hip_runtime.h>
#include <hip/hip_bf16.h>

typedef __attribute__((ext_vector_type(16))) _Float16 v16h;
typedef __attribute__((ext_vector_type(8)))  float    v8f;

#define BATCH   8
#define CCH     128
#define HD      32            // downsampled grid (32x32)
#define L       1024          // HD*HD
#define PDIM    1152          // C*9
#define CKDIM   2048          // C*16
#define ATT_SCALE 10.0f
#define EPSV    1e-4f

// ---- WMMA GEMM tiling: 128(M)x256(N) block, K-step 32, 8 waves (2x4), 64x64/wave ----
#define BM 128
#define BN 256
#define BK 32
#define LDSH 40               // BK + 8 halves padding (keeps 16B alignment, kills conflicts)

union FragLd { v16h v; uint4 q[2]; };

// ---------------- mask: mm[l] = 1 if all 3x3 neighborhood of downsampled mask is 0 ----
__global__ __launch_bounds__(1024) void mm_kernel(const float* __restrict__ mask,
                                                  float* __restrict__ MM) {
  int l = threadIdx.x;
  int i = l >> 5, j = l & 31;
  float s = 0.f;
  #pragma unroll
  for (int kh = 0; kh < 3; ++kh)
    #pragma unroll
    for (int kw = 0; kw < 3; ++kw) {
      int r = i + kh - 1, q = j + kw - 1;
      if ((unsigned)r < 32u && (unsigned)q < 32u)
        s += mask[(size_t)(16 * r) * 512 + 16 * q];   // mask[0,0,::16,::16]
    }
  MM[l] = (s == 0.f) ? 1.f : 0.f;
}

// ---------------- WN[b][l][p] = normalized 3x3 patches of b_ds, f16 -----------------
__global__ __launch_bounds__(128) void prep_wn(const float* __restrict__ B,
                                               _Float16* __restrict__ WN) {
  __shared__ float red[128];
  int blk = blockIdx.x;
  int b = blk >> 10, l = blk & 1023;
  int i = l >> 5, j = l & 31;
  int c = threadIdx.x;
  float w[9]; float ss = 0.f;
  #pragma unroll
  for (int kh = 0; kh < 3; ++kh)
    #pragma unroll
    for (int kw = 0; kw < 3; ++kw) {
      int r = i + kh - 1, q = j + kw - 1;
      float v = 0.f;
      if ((unsigned)r < 32u && (unsigned)q < 32u)
        v = B[(((size_t)b * CCH + c) * 64 + 2 * r) * 64 + 2 * q];
      w[kh * 3 + kw] = v; ss += v * v;
    }
  red[c] = ss; __syncthreads();
  for (int o = 64; o > 0; o >>= 1) { if (c < o) red[c] += red[c + o]; __syncthreads(); }
  float inv = rsqrtf(red[0] + (float)PDIM * EPSV);   // denom = sqrt(sum(w*w + EPS))
  _Float16* dst = WN + ((size_t)(b * L + l)) * PDIM + c * 9;
  #pragma unroll
  for (int t = 0; t < 9; ++t) dst[t] = (_Float16)(w[t] * inv);
}

// ---------------- FPT[b][hw][p] = 3x3 patches of f_ds (A matrix of GEMM1), f16 ------
__global__ __launch_bounds__(256) void prep_fpt(const float* __restrict__ F,
                                                _Float16* __restrict__ FPT) {
  int idx = blockIdx.x * 256 + threadIdx.x;          // 8*1024*1152 total
  int p = idx % PDIM;
  int rest = idx / PDIM;
  int hw = rest & 1023, b = rest >> 10;
  int c = p / 9, t = p % 9, kh = t / 3, kw = t % 3;
  int h = hw >> 5, w = hw & 31;
  int r = h + kh - 1, q = w + kw - 1;
  float v = 0.f;
  if ((unsigned)r < 32u && (unsigned)q < 32u)
    v = F[(((size_t)b * CCH + c) * 64 + 2 * r) * 64 + 2 * q];
  FPT[idx] = (_Float16)v;
}

// ------------- RWT[b][c*16+kh*4+kw][l] = raw 4x4 stride-2 patches of b, f16 ---------
__global__ __launch_bounds__(256) void prep_rwt(const float* __restrict__ B,
                                                _Float16* __restrict__ RWT) {
  int idx = blockIdx.x * 256 + threadIdx.x;          // 8*2048*1024 total
  int l = idx & 1023;
  int ck = (idx >> 10) & 2047;
  int b = idx >> 21;
  int c = ck >> 4, t = ck & 15, kh = t >> 2, kw = t & 3;
  int i = l >> 5, j = l & 31;
  int row = 2 * i - 1 + kh, col = 2 * j - 1 + kw;
  float v = 0.f;
  if ((unsigned)row < 64u && (unsigned)col < 64u)
    v = B[(((size_t)b * CCH + c) * 64 + row) * 64 + col];
  RWT[idx] = (_Float16)v;
}

// ---------------- shared WMMA GEMM: D[M][N] = A[M][K] * Bt[N][K]^T ------------------
// mode 0: store f32 row-major into Cout (per-batch M*N)
// mode 1: patch overlap-add: M=hw(i,j), N=(c,kh,kw) -> atomicAdd Yout[b,c,2i-1+kh,2j-1+kw] += 0.25*v
__global__ __launch_bounds__(256)
void gemm_wmma(const _Float16* __restrict__ A, const _Float16* __restrict__ Bt,
               float* __restrict__ Cout, int M, int N, int K,
               int mode, float* __restrict__ Yout) {
  __shared__ _Float16 Ash[BM * LDSH];                // 128x32 (+pad) = 10 KB
  __shared__ _Float16 Bsh[BN * LDSH];                // 256x32 (+pad) = 20 KB
  const int b = blockIdx.z;
  const int mBase = blockIdx.y * BM;
  const int nBase = blockIdx.x * BN;
  const _Float16* Ab = A + (size_t)b * M * K;
  const _Float16* Bb = Bt + (size_t)b * N * K;
  const int tid = threadIdx.x;
  const int lane = tid & 31;
  const int wave = tid >> 5;
  const int wm = wave & 1;                  // 2 waves over M
  const int wn = wave >> 1;                 // 4 waves over N
  const int laneLo = lane & 15;
  const int laneHi = lane >> 4;
  v8f acc[4][4] = {};                       // 64x64 per wave: 4x4 16x16 fragments

  for (int k0 = 0; k0 < K; k0 += BK) {
    __syncthreads();
    // stage A tile 128x32: 16B per thread x2 reps (coalesced)
    #pragma unroll
    for (int rep = 0; rep < 2; ++rep) {
      int idx = tid + rep * 256;            // 0..511
      int row = idx >> 2;
      int col = (idx & 3) * 8;
      *(uint4*)&Ash[row * LDSH + col] =
          *(const uint4*)&Ab[(size_t)(mBase + row) * K + k0 + col];
    }
    // stage Bt tile 256x32: 16B per thread x4 reps
    #pragma unroll
    for (int rep = 0; rep < 4; ++rep) {
      int idx = tid + rep * 256;            // 0..1023
      int row = idx >> 2;
      int col = (idx & 3) * 8;
      *(uint4*)&Bsh[row * LDSH + col] =
          *(const uint4*)&Bb[(size_t)(nBase + row) * K + k0 + col];
    }
    __syncthreads();
    // A frag: lanes 0-15 -> M rows, K 0-7 & 16-23; lanes 16-31 -> K 8-15 & 24-31
    FragLd af[4], bf[4];
    #pragma unroll
    for (int sm = 0; sm < 4; ++sm) {
      int r = wm * 64 + sm * 16 + laneLo;
      int kb = laneHi * 8;
      af[sm].q[0] = *(const uint4*)&Ash[r * LDSH + kb];
      af[sm].q[1] = *(const uint4*)&Ash[r * LDSH + kb + 16];
    }
    // B frag: lanes 0-15 -> N cols, K 0-7 & 16-23; lanes 16-31 -> K 8-15 & 24-31
    #pragma unroll
    for (int sn = 0; sn < 4; ++sn) {
      int n = wn * 64 + sn * 16 + laneLo;
      int kb = laneHi * 8;
      bf[sn].q[0] = *(const uint4*)&Bsh[n * LDSH + kb];
      bf[sn].q[1] = *(const uint4*)&Bsh[n * LDSH + kb + 16];
    }
    #pragma unroll
    for (int sm = 0; sm < 4; ++sm)
      #pragma unroll
      for (int sn = 0; sn < 4; ++sn)
        acc[sm][sn] = __builtin_amdgcn_wmma_f32_16x16x32_f16(
            false, af[sm].v, false, bf[sn].v, (short)0, acc[sm][sn], false, false);
  }

  // D layout: VGPR r, lane -> (M = r + 8*(lane>>4), N = lane&15)
  #pragma unroll
  for (int sm = 0; sm < 4; ++sm)
    #pragma unroll
    for (int sn = 0; sn < 4; ++sn)
      #pragma unroll
      for (int r = 0; r < 8; ++r) {
        int mIdx = mBase + wm * 64 + sm * 16 + r + laneHi * 8;
        int nIdx = nBase + wn * 64 + sn * 16 + laneLo;
        float v = acc[sm][sn][r];
        if (mode == 0) {
          Cout[(size_t)b * M * N + (size_t)mIdx * N + nIdx] = v;
        } else {
          int i = mIdx >> 5, j = mIdx & 31;
          int c = nIdx >> 4, t = nIdx & 15, kh = t >> 2, kw = t & 3;
          int ho = 2 * i - 1 + kh, wo = 2 * j - 1 + kw;
          if ((unsigned)ho < 64u && (unsigned)wo < 64u)
            unsafeAtomicAdd(&Yout[(((size_t)b * CCH + c) * 64 + ho) * 64 + wo], v * 0.25f);
        }
      }
}

// -------- fused masked softmax over l (row-contiguous), argmax -> offsets, f16 cast ----
__global__ __launch_bounds__(256)
void softmax_argmax(const float* __restrict__ YIT, const float* __restrict__ MM,
                    _Float16* __restrict__ YIST, float* __restrict__ outOff) {
  __shared__ float sval[256];
  __shared__ int   sidx[256];
  const int row = blockIdx.x;               // b*1024 + hw
  const int tid = threadIdx.x;
  const float* src = YIT + (size_t)row * L;
  float s[4]; float mloc = -3.0e38f;
  #pragma unroll
  for (int u = 0; u < 4; ++u) {
    int l = tid + u * 256;
    s[u] = src[l] * MM[l] * ATT_SCALE;
    mloc = fmaxf(mloc, s[u]);
  }
  sval[tid] = mloc; __syncthreads();
  for (int o = 128; o > 0; o >>= 1) { if (tid < o) sval[tid] = fmaxf(sval[tid], sval[tid + o]); __syncthreads(); }
  float mx = sval[0]; __syncthreads();
  float e[4]; float sum = 0.f;
  #pragma unroll
  for (int u = 0; u < 4; ++u) { e[u] = __expf(s[u] - mx); sum += e[u]; }
  sval[tid] = sum; __syncthreads();
  for (int o = 128; o > 0; o >>= 1) { if (tid < o) sval[tid] += sval[tid + o]; __syncthreads(); }
  float inv = 1.f / sval[0]; __syncthreads();
  float bv = -1.f; int bi = 0;
  #pragma unroll
  for (int u = 0; u < 4; ++u) {
    int l = tid + u * 256;
    float o = e[u] * inv * MM[l];
    YIST[(size_t)row * L + l] = (_Float16)o;
    if (o > bv) { bv = o; bi = l; }
  }
  sval[tid] = bv; sidx[tid] = bi; __syncthreads();
  for (int o = 128; o > 0; o >>= 1) {
    if (tid < o) {
      float ov = sval[tid + o]; int oi = sidx[tid + o];
      if (ov > sval[tid] || (ov == sval[tid] && oi < sidx[tid])) { sval[tid] = ov; sidx[tid] = oi; }
    }
    __syncthreads();
  }
  if (tid == 0) {
    int b = row >> 10, hw = row & 1023;
    int h = hw >> 5, w = hw & 31, off = sidx[0];
    outOff[(((size_t)b * 2 + 0) * HD + h) * HD + w] = (float)(off >> 5);
    outOff[(((size_t)b * 2 + 1) * HD + h) * HD + w] = (float)(off & 31);
  }
}

extern "C" void kernel_launch(void* const* d_in, const int* in_sizes, int n_in,
                              void* d_out, int out_size, void* d_ws, size_t ws_size,
                              hipStream_t stream) {
  const float* f    = (const float*)d_in[0];   // (8,128,64,64)
  const float* b    = (const float*)d_in[1];   // (8,128,64,64)
  const float* mask = (const float*)d_in[2];   // (8,1,512,512) (only batch 0 used)
  float* out = (float*)d_out;                  // y (4194304 f32) ++ offsets (16384)

  char* ws = (char*)d_ws;
  size_t off = 0;
  auto carve = [&](size_t bytes) -> void* {
    off = (off + 255) & ~(size_t)255;
    void* p = ws + off; off += bytes; return p;
  };
  _Float16* WN   = (_Float16*)carve((size_t)BATCH * L * PDIM * 2);     // 18.9 MB
  _Float16* FPT  = (_Float16*)carve((size_t)BATCH * L * PDIM * 2);     // 18.9 MB
  _Float16* RWT  = (_Float16*)carve((size_t)BATCH * CKDIM * L * 2);    // 33.6 MB
  _Float16* YIST = (_Float16*)carve((size_t)BATCH * L * L * 2);        // 16.8 MB
  float*    YIT  = (float*)   carve((size_t)BATCH * L * L * 4);        // 33.6 MB
  float*    MM   = (float*)   carve((size_t)L * 4);                    // ~122 MB total

  // zero output (y region is accumulated via atomics)
  hipMemsetAsync(d_out, 0, (size_t)out_size * sizeof(float), stream);

  mm_kernel<<<1, 1024, 0, stream>>>(mask, MM);
  prep_wn  <<<BATCH * L, 128, 0, stream>>>(b, WN);
  prep_fpt <<<(BATCH * L * PDIM) / 256, 256, 0, stream>>>(f, FPT);
  prep_rwt <<<(BATCH * CKDIM * L) / 256, 256, 0, stream>>>(b, RWT);

  // GEMM1: yi^T[hw][l] = FPT[hw][p] * WN[l][p]^T   (M=1024, N=1024, K=1152)
  gemm_wmma<<<dim3(L / BN, L / BM, BATCH), 256, 0, stream>>>(
      FPT, WN, YIT, L, L, PDIM, 0, nullptr);

  softmax_argmax<<<BATCH * L, 256, 0, stream>>>(YIT, MM, YIST, out + (size_t)BATCH * CCH * 64 * 64);

  // GEMM2 + paste: Z[hw][ck] = YIST[hw][l] * RWT[ck][l]^T, scatter-add *0.25 into y
  gemm_wmma<<<dim3(CKDIM / BN, L / BM, BATCH), 256, 0, stream>>>(
      YIST, RWT, nullptr, L, CKDIM, L, 1, out);
}
